// ResidualMambaBlock_89077621719165
// MI455X (gfx1250) — matrix operand order repaired and checked by reading, not compiled, not run on hardware
//
#include <hip/hip_runtime.h>
#include <hip/hip_bf16.h>

// ---------------------------------------------------------------------------
// Mamba2 block on gfx1250 (MI455X):
//   - bf16 WMMA GEMMs, register-blocked 32x64 per wave (8 accumulators)
//   - software-pipelined register-resident selective scan
// ---------------------------------------------------------------------------

#define D_MODEL    512
#define D_STATE    128
#define D_CONV     5
#define HEADDIM    64
#define D_INNER    1024
#define NHEADS     16
#define CONV_DIM   1280          // D_INNER + 2*D_STATE
#define D_IN_PROJ  2320          // 2*D_INNER + 2*D_STATE + NHEADS
#define N_PROJ_PAD 2368          // padded to a multiple of 64 for 2x4 tile groups
#define SEQLEN     2048
#define EPSF       1e-5f

typedef __attribute__((ext_vector_type(16))) __bf16 v16bf;
typedef __attribute__((ext_vector_type(8)))  __bf16 v8bf;
typedef __attribute__((ext_vector_type(8)))  float  v8f;

// ---------------- fp32 -> bf16 conversion with zero padding ----------------
__global__ void cvt_f32_bf16_pad(const float* __restrict__ in, __bf16* __restrict__ out,
                                 int n_src, int n_dst) {
    int i = blockIdx.x * blockDim.x + threadIdx.x;
    if (i < n_dst) out[i] = (i < n_src) ? (__bf16)in[i] : (__bf16)0.0f;
}

// ---------------- LayerNorm over D_MODEL, emit bf16 ----------------
__global__ void layernorm_bf16(const float* __restrict__ x,
                               const float* __restrict__ w,
                               const float* __restrict__ b,
                               __bf16* __restrict__ xnb) {
    int l = blockIdx.x, t = threadIdx.x;            // 256 threads, 2 elems each
    const float* row = x + (size_t)l * D_MODEL;
    float x0 = row[t], x1 = row[t + 256];
    __shared__ float r1[256], r2[256];
    r1[t] = x0 + x1;
    r2[t] = x0 * x0 + x1 * x1;
    __syncthreads();
    for (int s = 128; s > 0; s >>= 1) {
        if (t < s) { r1[t] += r1[t + s]; r2[t] += r2[t + s]; }
        __syncthreads();
    }
    float mu  = r1[0] * (1.0f / D_MODEL);
    float var = r2[0] * (1.0f / D_MODEL) - mu * mu;
    float inv = rsqrtf(var + EPSF);
    __bf16* orow = xnb + (size_t)l * D_MODEL;
    orow[t]       = (__bf16)((x0 - mu) * inv * w[t]       + b[t]);
    orow[t + 256] = (__bf16)((x1 - mu) * inv * w[t + 256] + b[t + 256]);
}

// ---------------- bf16 WMMA GEMM: C[M,N] = A[M,K] * B[N,K]^T (+resid) ------
// Register-blocked: one wave computes a 32x64 tile block (2 M-tiles x 4
// N-tiles = 8 v8f accumulators), so each k-step does 8 WMMAs per 12 b128
// loads instead of 1 per 4.  M % 32 == 0, N % 64 == 0 required.
__global__ void wmma_gemm_bf16(const __bf16* __restrict__ A,
                               const __bf16* __restrict__ B,
                               const float*  __restrict__ resid,
                               float* __restrict__ C,
                               int M, int N, int K) {
    int lane = threadIdx.x;                               // 0..31
    int g = blockIdx.x * blockDim.y + threadIdx.y;        // wave -> tile group
    int groupsN = N >> 6;
    if (g >= (M >> 5) * groupsN) return;
    int gm = g / groupsN, gn = g % groupsN;

    int lo16 = lane & 15;
    int hi   = lane >> 4;                                 // 0 or 1

    // A: lanes hold rows; 16 contiguous K per lane starting at hi*16
    const __bf16* arow0 = A + (size_t)((gm << 5) + lo16) * K + (hi << 4);
    const __bf16* arow1 = arow0 + (size_t)16 * K;
    // B: lane = output column; 16 contiguous K per lane starting at hi*16
    const __bf16* brow0 = B + (size_t)((gn << 6) + lo16) * K + (hi << 4);

    v8f acc[8];
    #pragma unroll
    for (int i = 0; i < 8; ++i) acc[i] = (v8f){};

    for (int k = 0; k < K; k += 32) {
        union { v16bf v; v8bf h[2]; } a0, a1;
        a0.h[0] = *(const v8bf*)(arow0 + k);
        a0.h[1] = *(const v8bf*)(arow0 + k + 8);
        a1.h[0] = *(const v8bf*)(arow1 + k);
        a1.h[1] = *(const v8bf*)(arow1 + k + 8);
        v16bf bv[4];
        #pragma unroll
        for (int j = 0; j < 4; ++j)
            bv[j] = *(const v16bf*)(brow0 + (size_t)(j << 4) * K + k);
        #pragma unroll
        for (int j = 0; j < 4; ++j)
            acc[j] = __builtin_amdgcn_wmma_f32_16x16x32_bf16(
                false, a0.v, false, bv[j], (short)0, acc[j], false, false);
        #pragma unroll
        for (int j = 0; j < 4; ++j)
            acc[4 + j] = __builtin_amdgcn_wmma_f32_16x16x32_bf16(
                false, a1.v, false, bv[j], (short)0, acc[4 + j], false, false);
    }

    // C/D layout: VGPR r -> row r (lanes 0-15) / r+8 (lanes 16-31), col = lane&15
    #pragma unroll
    for (int ti = 0; ti < 8; ++ti) {
        int outm = (gm << 5) + ((ti >> 2) << 4) + (hi << 3);
        int outn = (gn << 6) + ((ti & 3) << 4) + lo16;
        #pragma unroll
        for (int r = 0; r < 8; ++r) {
            size_t idx = (size_t)(outm + r) * N + outn;
            float v = acc[ti][r];
            if (resid) v += resid[idx];
            C[idx] = v;
        }
    }
}

// ---------------- depthwise causal conv (k=5) + SiLU ----------------
__global__ void conv_silu(const float* __restrict__ zx,
                          const float* __restrict__ w,
                          const float* __restrict__ bias,
                          float* __restrict__ xbc) {
    int i = blockIdx.x * blockDim.x + threadIdx.x;        // over SEQLEN*CONV_DIM
    int l = i / CONV_DIM, c = i % CONV_DIM;
    float acc = bias[c];
    #pragma unroll
    for (int j = 0; j < D_CONV; ++j) {
        int ll = l - (D_CONV - 1) + j;
        if (ll >= 0)
            acc += w[c * D_CONV + j] * zx[(size_t)ll * N_PROJ_PAD + D_INNER + c];
    }
    xbc[i] = acc / (1.0f + expf(-acc));                   // silu
}

// ---------------- dt = softplus(dt_raw + bias); dA = exp(dt * -exp(A_log)) --
__global__ void dt_kernel(const float* __restrict__ zx,
                          const float* __restrict__ dt_bias,
                          const float* __restrict__ A_log,
                          float* __restrict__ dt_out,
                          float* __restrict__ dA_out) {
    int i = blockIdx.x * blockDim.x + threadIdx.x;        // over SEQLEN*NHEADS
    int l = i >> 4, h = i & 15;
    float s  = zx[(size_t)l * N_PROJ_PAD + (D_INNER + CONV_DIM) + h] + dt_bias[h];
    float dt = (s > 20.0f) ? s : log1pf(expf(s));
    dt_out[i] = dt;
    dA_out[i] = expf(dt * (-expf(A_log[h])));
}

// ---------------- selective scan: 1 block per head, state in registers -----
// 256 threads: thread t owns p = t>>2, state slice n in [(t&3)*32, +32).
// Software pipelined: global loads for step l+1 issue while computing step l
// from LDS; one barrier per step.
__global__ void scan_kernel(const float* __restrict__ xbc,
                            const float* __restrict__ dt,
                            const float* __restrict__ dA,
                            const float* __restrict__ Dp,
                            float* __restrict__ y) {
    __shared__ float sx[2][HEADDIM], sB[2][D_STATE], sC[2][D_STATE];
    __shared__ float sDt[2], sDa[2];
    int head = blockIdx.x, t = threadIdx.x;
    int p = t >> 2, nb = (t & 3) << 5;

    float h[32];
    #pragma unroll
    for (int i = 0; i < 32; ++i) h[i] = 0.0f;
    float Dh = Dp[head];

    // prologue: load step 0 and publish to buffer 0
    float r0 = 0.0f, r1 = 0.0f, rdt = 0.0f, rda = 0.0f;
    {
        const float* row = xbc;
        if (t < 64)       { r0 = row[head * HEADDIM + t];
                            r1 = row[D_INNER + D_STATE + 64 + t]; }
        else if (t < 192) { r0 = row[D_INNER + (t - 64)]; }
        else              { r0 = row[D_INNER + D_STATE + (t - 192)]; }
        if (t == 0) { rdt = dt[head]; rda = dA[head]; }
        if (t < 64)       { sx[0][t] = r0; sC[0][64 + t] = r1; }
        else if (t < 192) { sB[0][t - 64] = r0; }
        else              { sC[0][t - 192] = r0; }
        if (t == 0) { sDt[0] = rdt; sDa[0] = rda; }
    }
    __syncthreads();

    int buf = 0;
    for (int l = 0; l < SEQLEN; ++l) {
        // issue loads for step l+1 (overlap with compute below)
        float n0 = 0.0f, n1 = 0.0f;
        if (l + 1 < SEQLEN) {
            const float* row = xbc + (size_t)(l + 1) * CONV_DIM;
            if (t < 64)       { n0 = row[head * HEADDIM + t];
                                n1 = row[D_INNER + D_STATE + 64 + t]; }
            else if (t < 192) { n0 = row[D_INNER + (t - 64)]; }
            else              { n0 = row[D_INNER + D_STATE + (t - 192)]; }
            if (t == 0) { rdt = dt[(l + 1) * NHEADS + head];
                          rda = dA[(l + 1) * NHEADS + head]; }
        }

        // compute step l from LDS buffer `buf`
        float a = sDa[buf], dtx = sDt[buf] * sx[buf][p], acc = 0.0f;
        #pragma unroll
        for (int i = 0; i < 32; ++i) {
            h[i] = fmaf(h[i], a, dtx * sB[buf][nb + i]);
            acc  = fmaf(h[i], sC[buf][nb + i], acc);
        }
        acc += __shfl_xor(acc, 1, 32);                    // reduce 4 lanes / p
        acc += __shfl_xor(acc, 2, 32);
        if ((t & 3) == 0)
            y[(size_t)l * D_INNER + head * HEADDIM + p] = acc + Dh * sx[buf][p];

        // publish step l+1 into the other buffer
        int nxt = buf ^ 1;
        if (l + 1 < SEQLEN) {
            if (t < 64)       { sx[nxt][t] = n0; sC[nxt][64 + t] = n1; }
            else if (t < 192) { sB[nxt][t - 64] = n0; }
            else              { sC[nxt][t - 192] = n0; }
            if (t == 0) { sDt[nxt] = rdt; sDa[nxt] = rda; }
        }
        __syncthreads();
        buf = nxt;
    }
}

// ---------------- gate with silu(z), RMSNorm, emit bf16 ----------------
__global__ void gate_rmsnorm_bf16(const float* __restrict__ y,
                                  const float* __restrict__ zx,
                                  const float* __restrict__ norm_w,
                                  __bf16* __restrict__ ynb) {
    int l = blockIdx.x, t = threadIdx.x;                  // 256 threads, 4 elems
    float v[4]; float ss = 0.0f;
    #pragma unroll
    for (int j = 0; j < 4; ++j) {
        int e = t + j * 256;
        float z = zx[(size_t)l * N_PROJ_PAD + e];
        float g = y[(size_t)l * D_INNER + e] * (z / (1.0f + expf(-z)));
        v[j] = g; ss += g * g;
    }
    __shared__ float red[256];
    red[t] = ss;
    __syncthreads();
    for (int s = 128; s > 0; s >>= 1) {
        if (t < s) red[t] += red[t + s];
        __syncthreads();
    }
    float scale = rsqrtf(red[0] * (1.0f / D_INNER) + EPSF);
    #pragma unroll
    for (int j = 0; j < 4; ++j) {
        int e = t + j * 256;
        ynb[(size_t)l * D_INNER + e] = (__bf16)(v[j] * scale * norm_w[e]);
    }
}

// ---------------------------------------------------------------------------
extern "C" void kernel_launch(void* const* d_in, const int* in_sizes, int n_in,
                              void* d_out, int out_size, void* d_ws, size_t ws_size,
                              hipStream_t stream) {
    const float* x       = (const float*)d_in[0];
    const float* ln_w    = (const float*)d_in[1];
    const float* ln_b    = (const float*)d_in[2];
    const float* Win     = (const float*)d_in[3];
    const float* conv_w  = (const float*)d_in[4];
    const float* conv_b  = (const float*)d_in[5];
    const float* dt_bias = (const float*)d_in[6];
    const float* A_log   = (const float*)d_in[7];
    const float* Dp      = (const float*)d_in[8];
    const float* norm_w  = (const float*)d_in[9];
    const float* Wout    = (const float*)d_in[10];
    float* out = (float*)d_out;

    char* w = (char*)d_ws;
    size_t off = 0;
    auto take = [&](size_t bytes) { char* p = w + off; off += (bytes + 255) & ~(size_t)255; return p; };
    __bf16* winb  = (__bf16*)take((size_t)N_PROJ_PAD * D_MODEL * 2);   // zero-padded rows
    __bf16* woutb = (__bf16*)take((size_t)D_MODEL * D_INNER * 2);
    __bf16* xnb   = (__bf16*)take((size_t)SEQLEN * D_MODEL * 2);
    float*  zx    = (float*) take((size_t)SEQLEN * N_PROJ_PAD * 4);
    float*  xbc   = (float*) take((size_t)SEQLEN * CONV_DIM * 4);
    float*  dtb   = (float*) take((size_t)SEQLEN * NHEADS * 4);
    float*  dab   = (float*) take((size_t)SEQLEN * NHEADS * 4);
    float*  yscan = (float*) take((size_t)SEQLEN * D_INNER * 4);
    __bf16* ynb   = (__bf16*)take((size_t)SEQLEN * D_INNER * 2);

    // 1) weight conversion to bf16 (Win padded with zero rows to N_PROJ_PAD)
    {
        int nsrc = D_IN_PROJ * D_MODEL, ndst = N_PROJ_PAD * D_MODEL;
        cvt_f32_bf16_pad<<<(ndst + 255) / 256, 256, 0, stream>>>(Win, winb, nsrc, ndst);
        int n2 = D_MODEL * D_INNER;
        cvt_f32_bf16_pad<<<(n2 + 255) / 256, 256, 0, stream>>>(Wout, woutb, n2, n2);
    }
    // 2) LayerNorm -> bf16 activations
    layernorm_bf16<<<SEQLEN, 256, 0, stream>>>(x, ln_w, ln_b, xnb);
    // 3) in-proj GEMM: zx[2048, 2368] = xn * Win_pad^T   (WMMA bf16, 2x4 blocked)
    {
        int groups = (SEQLEN / 32) * (N_PROJ_PAD / 64);   // 64*37 = 2368 waves
        wmma_gemm_bf16<<<groups / 4, dim3(32, 4), 0, stream>>>(
            xnb, winb, nullptr, zx, SEQLEN, N_PROJ_PAD, D_MODEL);
    }
    // 4) depthwise conv + silu
    conv_silu<<<(SEQLEN * CONV_DIM) / 256, 256, 0, stream>>>(zx, conv_w, conv_b, xbc);
    // 5) dt / dA
    dt_kernel<<<(SEQLEN * NHEADS) / 256, 256, 0, stream>>>(zx, dt_bias, A_log, dtb, dab);
    // 6) selective scan (+ D skip), software pipelined
    scan_kernel<<<NHEADS, 256, 0, stream>>>(xbc, dtb, dab, Dp, yscan);
    // 7) gate + RMSNorm -> bf16
    gate_rmsnorm_bf16<<<SEQLEN, 256, 0, stream>>>(yscan, zx, norm_w, ynb);
    // 8) out-proj GEMM + residual: out = yn * Wout^T + x  (WMMA bf16, 2x4 blocked)
    {
        int groups = (SEQLEN / 32) * (D_MODEL / 64);      // 64*8 = 512 waves
        wmma_gemm_bf16<<<groups / 4, dim3(32, 4), 0, stream>>>(
            ynb, woutb, x, out, SEQLEN, D_MODEL, D_INNER);
    }
}
